// BiHgru2_1d_31636729102919
// MI455X (gfx1250) — matrix operand order repaired and checked by reading, not compiled, not run on hardware
//
#include <hip/hip_runtime.h>

typedef unsigned short u16;
typedef unsigned int   u32;
typedef __bf16 bf16_t;
typedef bf16_t v16bf __attribute__((ext_vector_type(16)));
typedef float  v8f   __attribute__((ext_vector_type(8)));
typedef u32    v4u   __attribute__((ext_vector_type(4)));
typedef int    v4i   __attribute__((ext_vector_type(4)));

#define N_SEQ   2048
#define BATCH   4
#define DMODEL  1024
#define M_ROWS  8192      // N_SEQ*BATCH
#define N1      3072
#define KDIM    1024
#define KITER   (KDIM / 32)
#define CCH     8192      // BATCH*2048 scan channels
#define CHUNKS  32
#define CLEN    64        // N_SEQ / CHUNKS

#if __has_builtin(__builtin_amdgcn_global_load_async_to_lds_b128) && \
    __has_builtin(__builtin_amdgcn_s_wait_asynccnt)
#define HAS_ASYNC 1
#define WAIT_ASYNC(n) __builtin_amdgcn_s_wait_asynccnt(n)
#else
#define HAS_ASYNC 0
#define WAIT_ASYNC(n)
#endif

__device__ __forceinline__ u16 f2bf(float f) {
  u32 u = __float_as_uint(f);
  u32 r = u + 0x7FFFu + ((u >> 16) & 1u);
  return (u16)(r >> 16);
}
__device__ __forceinline__ float sigmoidf(float v) {
  return 1.0f / (1.0f + __expf(-v));
}

union FragU { v4u q[2]; v16bf v; };

// A-operand frag (16x32 bf16) from a row-major buffer: contiguous 16 K values
__device__ __forceinline__ v16bf fragA(const u16* p) {
  FragU f; f.q[0] = *(const v4u*)p; f.q[1] = *(const v4u*)(p + 8); return f.v;
}
// B-operand frag (32x16 bf16) from N-major buffer: contiguous 16 K values
__device__ __forceinline__ v16bf fragB(const u16* p) {
  FragU f; f.q[0] = *(const v4u*)p; f.q[1] = *(const v4u*)(p + 8); return f.v;
}

// 16-byte global -> LDS copy: async (ASYNCcnt) on gfx1250 toolchains that expose it
__device__ __forceinline__ void copy16(const u16* g, u16* l) {
#if HAS_ASYNC
  __builtin_amdgcn_global_load_async_to_lds_b128(
      (__attribute__((address_space(1))) v4i*)(void*)g,
      (__attribute__((address_space(3))) v4i*)(void*)l, 0, 0);
#else
  *(v4u*)l = *(const v4u*)g;
#endif
}

// stage kk of A(2 chunks)+B(1 chunk) into LDS buffer `buf` (names resolve in kernel scope)
#define ISSUE(kk, buf)                                      \
  do {                                                      \
    int kb_ = (kk) * 32;                                    \
    copy16(gA0 + kb_, &Abuf[buf][arow0][ac4]);              \
    copy16(gA1 + kb_, &Abuf[buf][arow1][ac4]);              \
    copy16(gB  + kb_, &Bbuf[buf][brow][bc4]);               \
  } while (0)

// ---------------- prologue: convert / transpose ----------------

__global__ void __launch_bounds__(256) cvt_x_kernel(const float* __restrict__ x, u16* __restrict__ xb) {
  int i = blockIdx.x * 256 + threadIdx.x;
  xb[i] = f2bf(x[i]);
}

// src: R x C fp32 row-major  ->  dst: C x R bf16 row-major
__global__ void __launch_bounds__(256) transpose_bf16(const float* __restrict__ src, u16* __restrict__ dst,
                                                      int R, int C) {
  __shared__ float tile[32][33];
  int tx = threadIdx.x, ty = threadIdx.y;
  int c = blockIdx.x * 32 + tx;
  #pragma unroll
  for (int i = ty; i < 32; i += 8) {
    int r = blockIdx.y * 32 + i;
    tile[i][tx] = src[(size_t)r * C + c];
  }
  __syncthreads();
  int r2 = blockIdx.y * 32 + tx;
  #pragma unroll
  for (int i = ty; i < 32; i += 8) {
    int c2 = blockIdx.x * 32 + i;
    dst[(size_t)c2 * R + r2] = f2bf(tile[tx][i]);
  }
}

// ---------------- GEMM1: feat = x @ W_in + b_in, fused activations ----------------
// block = 256 thr (8 waves) -> 128x64 C-tile; waves 4x2 of 32x32; LDS double-buffered async staging

__global__ void __launch_bounds__(256) gemm1_act(const u16* __restrict__ A, const u16* __restrict__ B,
                                                 const float* __restrict__ b_in, const float* __restrict__ lb,
                                                 float* __restrict__ inp, float* __restrict__ og,
                                                 float* __restrict__ lam) {
  __shared__ __align__(16) u16 Abuf[2][128][40];   // 32 K x 128 M, pitch 40 (80B, conflict-free b128)
  __shared__ __align__(16) u16 Bbuf[2][64][40];
  int tt = threadIdx.x;
  int lane = tt & 31;
  int w = tt >> 5;
  int wm = w >> 1, wn = w & 1;
  int m0 = (blockIdx.x / (N1 / 64)) * 128;
  int n0 = (blockIdx.x % (N1 / 64)) * 64;

  int arow0 = tt >> 2, ac4 = (tt & 3) * 8;
  int arow1 = arow0 + 64;
  int brow  = tt >> 2, bc4 = (tt & 3) * 8;
  const u16* gA0 = A + (size_t)(m0 + arow0) * KDIM + ac4;
  const u16* gA1 = A + (size_t)(m0 + arow1) * KDIM + ac4;
  const u16* gB  = B + (size_t)(n0 + brow)  * KDIM + bc4;

  int fr  = lane & 15;
  int fka = (lane >> 4) << 4;    // A frag K base within tile: 0/16
  int fkb = (lane >> 4) << 4;    // B frag K base within tile: 0/16

  v8f acc[2][2] = {{{}, {}}, {{}, {}}};
  ISSUE(0, 0);
  for (int kk = 0; kk < KITER; ++kk) {
    int cur = kk & 1;
    if (kk + 1 < KITER) { ISSUE(kk + 1, cur ^ 1); WAIT_ASYNC(3); }
    else                { WAIT_ASYNC(0); }
    __syncthreads();
    v16bf a0 = fragA(&Abuf[cur][wm * 32 + fr][fka]);
    v16bf a1 = fragA(&Abuf[cur][wm * 32 + 16 + fr][fka]);
    v16bf b0 = fragB(&Bbuf[cur][wn * 32 + fr][fkb]);
    v16bf b1 = fragB(&Bbuf[cur][wn * 32 + 16 + fr][fkb]);
    acc[0][0] = __builtin_amdgcn_wmma_f32_16x16x32_bf16(false, a0, false, b0, (short)0, acc[0][0], false, false);
    acc[0][1] = __builtin_amdgcn_wmma_f32_16x16x32_bf16(false, a0, false, b1, (short)0, acc[0][1], false, false);
    acc[1][0] = __builtin_amdgcn_wmma_f32_16x16x32_bf16(false, a1, false, b0, (short)0, acc[1][0], false, false);
    acc[1][1] = __builtin_amdgcn_wmma_f32_16x16x32_bf16(false, a1, false, b1, (short)0, acc[1][1], false, false);
    __syncthreads();
  }

  int mlane = (lane >> 4) << 3;
  int nlane = lane & 15;
  #pragma unroll
  for (int mi = 0; mi < 2; ++mi) {
    #pragma unroll
    for (int ni = 0; ni < 2; ++ni) {
      #pragma unroll
      for (int r = 0; r < 8; ++r) {
        int m    = m0 + wm * 32 + mi * 16 + mlane + r;
        int ncol = n0 + wn * 32 + ni * 16 + nlane;
        float v = acc[mi][ni][r] + b_in[ncol];
        int third = ncol >> 10;
        int c = ncol & 1023;
        size_t o = (size_t)m * 1024 + c;
        if (third == 0) {
          inp[o] = v * sigmoidf(v);                  // silu
        } else if (third == 1) {
          og[o] = sigmoidf(v);
        } else {
          float l = lb[c];
          lam[o] = l + (1.0f - l) * sigmoidf(v);
        }
      }
    }
  }
}

// ---------------- chunked bidirectional scan ----------------
// channel c in [0,8192): c = bb*2048 + cc, cc = hh*4 + dd*2 + ee
// a_t = lam[t,bb,hh*2+dd], u_t = (1-a_t) * inp[t,bb,hh*2+ee]

__global__ void __launch_bounds__(256) scan_carry(const float* __restrict__ lam, const float* __restrict__ inp,
                                                  float* __restrict__ Aar, float* __restrict__ Sf,
                                                  float* __restrict__ Sr) {
  int tid = blockIdx.x * 256 + threadIdx.x;        // CHUNKS*CCH
  int c = tid & (CCH - 1);
  int k = tid >> 13;
  int bb = c >> 11;
  int cc = c & 2047;
  int hh = cc >> 2, dd = (cc >> 1) & 1, ee = cc & 1;
  int li = hh * 2 + dd, ii = hh * 2 + ee;
  float P = 1.0f, sf = 0.0f, sr = 0.0f;
  int t0 = k * CLEN;
  for (int t = t0; t < t0 + CLEN; ++t) {
    size_t row = (size_t)(t * BATCH + bb) * 1024;
    float a = lam[row + li];
    float u = (1.0f - a) * inp[row + ii];
    sf = a * sf + u;    // forward local final
    sr += P * u;        // reverse value at chunk-left (ascending form)
    P *= a;             // running product of a
  }
  Aar[tid] = P; Sf[tid] = sf; Sr[tid] = sr;
}

__global__ void __launch_bounds__(256) scan_combine(const float* __restrict__ Aar, const float* __restrict__ Sf,
                                                    const float* __restrict__ Sr, float* __restrict__ Inf_,
                                                    float* __restrict__ Inr_) {
  int c = blockIdx.x * 256 + threadIdx.x;          // CCH threads
  float s = 0.0f;
  for (int k = 0; k < CHUNKS; ++k) {
    int i = k * CCH + c;
    Inf_[i] = s;
    s = Aar[i] * s + Sf[i];
  }
  s = 0.0f;
  for (int k = CHUNKS - 1; k >= 0; --k) {
    int i = k * CCH + c;
    Inr_[i] = s;
    s = Aar[i] * s + Sr[i];
  }
}

__global__ void __launch_bounds__(256) scan_fwd(const float* __restrict__ lam, const float* __restrict__ inp,
                                                const float* __restrict__ Inf_, float* __restrict__ stf) {
  int tid = blockIdx.x * 256 + threadIdx.x;        // CHUNKS*CCH
  int c = tid & (CCH - 1);
  int k = tid >> 13;
  int bb = c >> 11;
  int cc = c & 2047;
  int hh = cc >> 2, dd = (cc >> 1) & 1, ee = cc & 1;
  int li = hh * 2 + dd, ii = hh * 2 + ee;
  float s = Inf_[tid];
  int t0 = k * CLEN;
  for (int t = t0; t < t0 + CLEN; ++t) {
    size_t row = (size_t)(t * BATCH + bb) * 1024;
    float a = lam[row + li];
    float u = (1.0f - a) * inp[row + ii];
    s = a * s + u;
    stf[(size_t)(t * BATCH + bb) * 2048 + cc] = s;
  }
}

// reverse fixup fused with E-contraction: y[t,bb,hh*2+ee] = sum_dd (fwd+rev)[hh,dd,ee] * og[hh*2+dd]
__global__ void __launch_bounds__(256) scan_rev_contract(const float* __restrict__ lam, const float* __restrict__ inp,
                                                         const float* __restrict__ og, const float* __restrict__ Inr_,
                                                         const float* __restrict__ stf, float* __restrict__ yraw) {
  int tid = blockIdx.x * 256 + threadIdx.x;        // CHUNKS * 4096
  int q = tid & 4095;
  int k = tid >> 12;
  int bb = q >> 10;
  int rest = q & 1023;
  int hh = rest >> 1, ee = rest & 1;
  int cc0 = hh * 4 + ee;        // dd=0
  int cc1 = cc0 + 2;            // dd=1
  int li0 = hh * 2, li1 = hh * 2 + 1, ii = hh * 2 + ee;
  float s0 = Inr_[k * CCH + bb * 2048 + cc0];
  float s1 = Inr_[k * CCH + bb * 2048 + cc1];
  int t0 = k * CLEN;
  for (int t = t0 + CLEN - 1; t >= t0; --t) {
    size_t r1 = (size_t)(t * BATCH + bb) * 1024;
    size_t r2 = (size_t)(t * BATCH + bb) * 2048;
    float ip = inp[r1 + ii];
    float a0 = lam[r1 + li0];
    float a1 = lam[r1 + li1];
    s0 = a0 * s0 + (1.0f - a0) * ip;
    s1 = a1 * s1 + (1.0f - a1) * ip;
    float tot0 = stf[r2 + cc0] + s0;
    float tot1 = stf[r2 + cc1] + s1;
    yraw[r1 + ii] = tot0 * og[r1 + li0] + tot1 * og[r1 + li1];
  }
}

// ---------------- layernorm -> bf16 ----------------

__global__ void __launch_bounds__(256) layernorm_bf16(const float* __restrict__ yraw,
                                                      const float* __restrict__ gamma,
                                                      const float* __restrict__ beta,
                                                      u16* __restrict__ yb) {
  __shared__ float sh[16];
  int row = blockIdx.x;
  const float* p = yraw + (size_t)row * 1024;
  int t = threadIdx.x;
  float4 v = *(const float4*)(p + t * 4);
  float s  = v.x + v.y + v.z + v.w;
  float ss = v.x * v.x + v.y * v.y + v.z * v.z + v.w * v.w;
  #pragma unroll
  for (int off = 16; off; off >>= 1) {
    s  += __shfl_xor(s,  off, 32);
    ss += __shfl_xor(ss, off, 32);
  }
  int w = t >> 5;
  if ((t & 31) == 0) { sh[w] = s; sh[8 + w] = ss; }
  __syncthreads();
  float st = 0.0f, sst = 0.0f;
  #pragma unroll
  for (int i = 0; i < 8; ++i) { st += sh[i]; sst += sh[8 + i]; }
  float mu  = st * (1.0f / 1024.0f);
  float var = sst * (1.0f / 1024.0f) - mu * mu;
  float rs  = rsqrtf(var + 1e-5f);
  int c0 = t * 4;
  u16* o = yb + (size_t)row * 1024 + c0;
  float vv[4] = {v.x, v.y, v.z, v.w};
  #pragma unroll
  for (int i = 0; i < 4; ++i)
    o[i] = f2bf((vv[i] - mu) * rs * gamma[c0 + i] + beta[c0 + i]);
}

// ---------------- GEMM2: out = y_norm @ W_out + b_out ----------------

__global__ void __launch_bounds__(256) gemm2_out(const u16* __restrict__ A, const u16* __restrict__ B,
                                                 const float* __restrict__ b_out, float* __restrict__ out) {
  __shared__ __align__(16) u16 Abuf[2][128][40];
  __shared__ __align__(16) u16 Bbuf[2][64][40];
  int tt = threadIdx.x;
  int lane = tt & 31;
  int w = tt >> 5;
  int wm = w >> 1, wn = w & 1;
  int m0 = (blockIdx.x / (DMODEL / 64)) * 128;
  int n0 = (blockIdx.x % (DMODEL / 64)) * 64;

  int arow0 = tt >> 2, ac4 = (tt & 3) * 8;
  int arow1 = arow0 + 64;
  int brow  = tt >> 2, bc4 = (tt & 3) * 8;
  const u16* gA0 = A + (size_t)(m0 + arow0) * KDIM + ac4;
  const u16* gA1 = A + (size_t)(m0 + arow1) * KDIM + ac4;
  const u16* gB  = B + (size_t)(n0 + brow)  * KDIM + bc4;

  int fr  = lane & 15;
  int fka = (lane >> 4) << 4;
  int fkb = (lane >> 4) << 4;

  v8f acc[2][2] = {{{}, {}}, {{}, {}}};
  ISSUE(0, 0);
  for (int kk = 0; kk < KITER; ++kk) {
    int cur = kk & 1;
    if (kk + 1 < KITER) { ISSUE(kk + 1, cur ^ 1); WAIT_ASYNC(3); }
    else                { WAIT_ASYNC(0); }
    __syncthreads();
    v16bf a0 = fragA(&Abuf[cur][wm * 32 + fr][fka]);
    v16bf a1 = fragA(&Abuf[cur][wm * 32 + 16 + fr][fka]);
    v16bf b0 = fragB(&Bbuf[cur][wn * 32 + fr][fkb]);
    v16bf b1 = fragB(&Bbuf[cur][wn * 32 + 16 + fr][fkb]);
    acc[0][0] = __builtin_amdgcn_wmma_f32_16x16x32_bf16(false, a0, false, b0, (short)0, acc[0][0], false, false);
    acc[0][1] = __builtin_amdgcn_wmma_f32_16x16x32_bf16(false, a0, false, b1, (short)0, acc[0][1], false, false);
    acc[1][0] = __builtin_amdgcn_wmma_f32_16x16x32_bf16(false, a1, false, b0, (short)0, acc[1][0], false, false);
    acc[1][1] = __builtin_amdgcn_wmma_f32_16x16x32_bf16(false, a1, false, b1, (short)0, acc[1][1], false, false);
    __syncthreads();
  }

  int mlane = (lane >> 4) << 3;
  int nlane = lane & 15;
  #pragma unroll
  for (int mi = 0; mi < 2; ++mi) {
    #pragma unroll
    for (int ni = 0; ni < 2; ++ni) {
      #pragma unroll
      for (int r = 0; r < 8; ++r) {
        int m    = m0 + wm * 32 + mi * 16 + mlane + r;
        int ncol = n0 + wn * 32 + ni * 16 + nlane;
        out[(size_t)m * 1024 + ncol] = acc[mi][ni][r] + b_out[ncol];
      }
    }
  }
}

// ---------------- launch ----------------

extern "C" void kernel_launch(void* const* d_in, const int* in_sizes, int n_in,
                              void* d_out, int out_size, void* d_ws, size_t ws_size,
                              hipStream_t stream) {
  (void)in_sizes; (void)n_in; (void)out_size; (void)ws_size;
  const float* x     = (const float*)d_in[0];
  const float* lb    = (const float*)d_in[1];
  const float* Win   = (const float*)d_in[2];
  const float* bin   = (const float*)d_in[3];
  const float* Wout  = (const float*)d_in[4];
  const float* bout  = (const float*)d_in[5];
  const float* gamma = (const float*)d_in[6];
  const float* beta  = (const float*)d_in[7];
  float* out = (float*)d_out;

  char* ws = (char*)d_ws;
  const size_t MB = 1ull << 20;
  float* inp   = (float*)(ws + 0 * MB);      // 32MB (M,1024) f32
  float* og    = (float*)(ws + 32 * MB);     // 32MB
  float* lam   = (float*)(ws + 64 * MB);     // 32MB
  float* stf   = (float*)(ws + 96 * MB);     // 64MB (M,2048) f32
  float* yraw  = (float*)(ws + 160 * MB);    // 32MB
  u16*   xb    = (u16*)  (ws + 192 * MB);    // 16MB (M,1024) bf16
  u16*   winT  = (u16*)  (ws + 208 * MB);    // 6MB  (3072,1024) bf16
  u16*   woutT = (u16*)  (ws + 214 * MB);    // 2MB  (1024,1024) bf16
  u16*   yb    = (u16*)  (ws + 216 * MB);    // 16MB (M,1024) bf16
  float* Aar   = (float*)(ws + 232 * MB);    // 1MB each
  float* Sf    = (float*)(ws + 233 * MB);
  float* Sr    = (float*)(ws + 234 * MB);
  float* Inf_  = (float*)(ws + 235 * MB);
  float* Inr_  = (float*)(ws + 236 * MB);

  dim3 tb(32, 8);
  cvt_x_kernel<<<(M_ROWS * DMODEL) / 256, 256, 0, stream>>>(x, xb);
  transpose_bf16<<<dim3(N1 / 32, DMODEL / 32), tb, 0, stream>>>(Win, winT, DMODEL, N1);
  transpose_bf16<<<dim3(DMODEL / 32, DMODEL / 32), tb, 0, stream>>>(Wout, woutT, DMODEL, DMODEL);

  gemm1_act<<<(M_ROWS / 128) * (N1 / 64), 256, 0, stream>>>(xb, winT, bin, lb, inp, og, lam);

  scan_carry<<<(CHUNKS * CCH) / 256, 256, 0, stream>>>(lam, inp, Aar, Sf, Sr);
  scan_combine<<<CCH / 256, 256, 0, stream>>>(Aar, Sf, Sr, Inf_, Inr_);
  scan_fwd<<<(CHUNKS * CCH) / 256, 256, 0, stream>>>(lam, inp, Inf_, stf);
  scan_rev_contract<<<(CHUNKS * 4096) / 256, 256, 0, stream>>>(lam, inp, og, Inr_, stf, yraw);

  layernorm_bf16<<<M_ROWS, 256, 0, stream>>>(yraw, gamma, beta, yb);
  gemm2_out<<<(M_ROWS / 128) * (DMODEL / 64), 256, 0, stream>>>(yb, woutT, bout, out);
}
